// Bottom2Up_Net_45260365365558
// MI455X (gfx1250) — matrix operand order repaired and checked by reading, not compiled, not run on hardware
//
#include <hip/hip_runtime.h>
#include <math.h>
#include <stdint.h>

// ---------------------------------------------------------------------------
// Problem constants (from the reference)
// ---------------------------------------------------------------------------
static constexpr int kB = 32, kL = 512, kN = 1024, kP = 32, kD = 256, kH = 256;
static constexpr float kFT  = 0.3f;
static constexpr float kEPS = 1e-30f;

typedef __attribute__((ext_vector_type(8)))  _Float16 v8h;
typedef __attribute__((ext_vector_type(16))) _Float16 v16h;
typedef __attribute__((ext_vector_type(8)))  float    v8f;
typedef int v4i __attribute__((vector_size(16)));   // matches builtin param type

union Frag16 { v16h v; v8h h[2]; };

// CDNA5 async global->LDS copy path (ASYNCcnt-tracked), with sync fallback.
#if defined(__gfx1250__) && __has_builtin(__builtin_amdgcn_global_load_async_to_lds_b128)
#define USE_ASYNC_COPY 1
#else
#define USE_ASYNC_COPY 0
#endif

#if USE_ASYNC_COPY
typedef __attribute__((address_space(1))) v4i* gv4i_p;
typedef __attribute__((address_space(3))) v4i* lv4i_p;
__device__ __forceinline__ void async_cp16(const _Float16* g, _Float16* l) {
  __builtin_amdgcn_global_load_async_to_lds_b128(
      (gv4i_p)g, (lv4i_p)l, 0, 0);
}
#if __has_builtin(__builtin_amdgcn_s_wait_asynccnt)
#define WAIT_ASYNC_LE(n) __builtin_amdgcn_s_wait_asynccnt(n)
#else
#define WAIT_ASYNC_LE(n) asm volatile("s_wait_asynccnt %0" ::"i"(n) : "memory")
#endif
#endif

// ---------------------------------------------------------------------------
// Generic WMMA GEMM:  C[M,256] (+)= A[M,K] @ Bt[256,K]^T  (+ bias[256])
//   A, Bt are f16 row-major (K contiguous).  C is f32.
//   Batched via blockIdx.z with element strides.  K % 32 == 0.
//   Block: 256 threads = 8 waves (wave32), tile 128x128, waves 2(M) x 4(N),
//   each wave: 4x2 fragments of 16x16, v_wmma_f32_16x16x32_f16.
//   Double-buffered LDS; global->LDS via async copies when available.
// ---------------------------------------------------------------------------
#define GBM 128
#define GBN 128
#define GBK 32
#define LDST 40   // halfs per LDS row (32 + 8 pad) -> 80B, keeps 16B alignment

__global__ __launch_bounds__(256) void wmma_gemm(
    const _Float16* __restrict__ A, int lda, long long strideA,
    const _Float16* __restrict__ Bt, int ldb, long long strideB,
    float* __restrict__ C, int ldc, long long strideC,
    const float* __restrict__ bias,
    int M, int K, int accumulate)
{
  __shared__ _Float16 As[2][GBM * LDST];
  __shared__ _Float16 Bs[2][GBN * LDST];

  A  += (long long)blockIdx.z * strideA;
  Bt += (long long)blockIdx.z * strideB;
  C  += (long long)blockIdx.z * strideC;

  const int tid   = threadIdx.x;
  const int lane  = tid & 31;
  const int wave  = tid >> 5;
  const int waveM = wave >> 2;   // 0..1 -> 64 rows each
  const int waveN = wave & 3;    // 0..3 -> 32 cols each
  const int ln    = lane & 15;
  const int lg    = lane >> 4;

  const int m0 = blockIdx.y * GBM;
  const int n0 = blockIdx.x * GBN;

  // ---- per-thread copy descriptors (2 A chunks + 2 B chunks of 16B) ----
  // Rows past M are clamped (never stored by the epilogue), keeping the
  // copy path branch-free and async-friendly.
  const int c0c   = tid;
  const int c1c   = tid + 256;
  const int rowA0 = c0c >> 2, offA0 = (c0c & 3) * 8;
  const int rowA1 = c1c >> 2, offA1 = (c1c & 3) * 8;
  const int gmA0  = min(m0 + rowA0, M - 1);
  const int gmA1  = min(m0 + rowA1, M - 1);
  const _Float16* gA0 = A + (long long)gmA0 * lda + offA0;
  const _Float16* gA1 = A + (long long)gmA1 * lda + offA1;
  const _Float16* gB0 = Bt + (long long)(n0 + rowA0) * ldb + offA0;
  const _Float16* gB1 = Bt + (long long)(n0 + rowA1) * ldb + offA1;
  const int lA0 = rowA0 * LDST + offA0;
  const int lA1 = rowA1 * LDST + offA1;

  v8f acc[4][2];
  for (int mf = 0; mf < 4; ++mf)
    for (int nf = 0; nf < 2; ++nf)
      for (int r = 0; r < 8; ++r) acc[mf][nf][r] = 0.0f;

  const int nK = K / GBK;

  auto issue_tile = [&](int kt, int buf) {
    const int k0 = kt * GBK;
#if USE_ASYNC_COPY
    async_cp16(gA0 + k0, &As[buf][lA0]);
    async_cp16(gA1 + k0, &As[buf][lA1]);
    async_cp16(gB0 + k0, &Bs[buf][lA0]);
    async_cp16(gB1 + k0, &Bs[buf][lA1]);
#else
    if (kt + 1 < nK) {  // emits global_prefetch_b8 on gfx1250
      __builtin_prefetch(gA0 + k0 + GBK, 0, 1);
      __builtin_prefetch(gB0 + k0 + GBK, 0, 1);
    }
    *(v8h*)(&As[buf][lA0]) = *(const v8h*)(gA0 + k0);
    *(v8h*)(&As[buf][lA1]) = *(const v8h*)(gA1 + k0);
    *(v8h*)(&Bs[buf][lA0]) = *(const v8h*)(gB0 + k0);
    *(v8h*)(&Bs[buf][lA1]) = *(const v8h*)(gB1 + k0);
#endif
  };

  issue_tile(0, 0);

  for (int kt = 0; kt < nK; ++kt) {
    const int cur = kt & 1;
    if (kt + 1 < nK) {
      issue_tile(kt + 1, cur ^ 1);   // overlap next tile's DMA with compute
#if USE_ASYNC_COPY
      WAIT_ASYNC_LE(4);              // tile kt landed; 4 ops of kt+1 in flight
#endif
    } else {
#if USE_ASYNC_COPY
      WAIT_ASYNC_LE(0);
#endif
    }
    __syncthreads();                 // tile kt visible to all waves

    // ---- fragments ----
    // A 16x32 f16 layout: lane (m=ln, g=lg): halfs0-7 = K in [8g,8g+8),
    //                     halfs8-15 = K in [16+8g, 16+8g+8)
    Frag16 af[4];
    #pragma unroll
    for (int mf = 0; mf < 4; ++mf) {
      const _Float16* p = &As[cur][(waveM * 64 + mf * 16 + ln) * LDST];
      af[mf].h[0] = *(const v8h*)(p + lg * 8);
      af[mf].h[1] = *(const v8h*)(p + 16 + lg * 8);
    }
    // B 32x16 f16 layout: lane (n=ln, g=lg): halfs0-7 = K in [8g,8g+8),
    //                     halfs8-15 = K in [16+8g, 16+8g+8)
    Frag16 bf[2];
    #pragma unroll
    for (int nf = 0; nf < 2; ++nf) {
      const _Float16* p = &Bs[cur][(waveN * 32 + nf * 16 + ln) * LDST + lg * 8];
      bf[nf].h[0] = *(const v8h*)(p);
      bf[nf].h[1] = *(const v8h*)(p + 16);
    }
    #pragma unroll
    for (int mf = 0; mf < 4; ++mf)
      #pragma unroll
      for (int nf = 0; nf < 2; ++nf)
        acc[mf][nf] = __builtin_amdgcn_wmma_f32_16x16x32_f16(
            false, af[mf].v, false, bf[nf].v,
            (short)0, acc[mf][nf], false, false);
    __syncthreads();                 // done reading buf cur before its reuse
  }

  // ---- epilogue: C layout row = 8*lg + r, col = ln within each 16x16 ----
  for (int mf = 0; mf < 4; ++mf) {
    for (int nf = 0; nf < 2; ++nf) {
      int col  = n0 + waveN * 32 + nf * 16 + ln;
      float bv = bias ? bias[col] : 0.0f;
      #pragma unroll
      for (int r = 0; r < 8; ++r) {
        int row = m0 + waveM * 64 + mf * 16 + lg * 8 + r;
        if (row < M) {
          long long idx = (long long)row * ldc + col;
          float v = acc[mf][nf][r] + bv;
          if (accumulate) v += C[idx];
          C[idx] = v;
        }
      }
    }
  }
}

// ---------------------------------------------------------------------------
// Small helpers
// ---------------------------------------------------------------------------
__device__ __forceinline__ float block_reduce_sum256(float v, float* sm) {
  int t = threadIdx.x;
  sm[t] = v; __syncthreads();
  for (int s = 128; s > 0; s >>= 1) {
    if (t < s) sm[t] += sm[t + s];
    __syncthreads();
  }
  float r = sm[0];
  __syncthreads();
  return r;
}

__global__ __launch_bounds__(256) void f32_to_f16(
    const float* __restrict__ in, _Float16* __restrict__ out, long long n)
{
  long long i = (long long)blockIdx.x * 256 + threadIdx.x;
  if (i < n) out[i] = (_Float16)in[i];
}

// awe = concat([word_outputs0, word_outputs1], axis=1) -> f16 [B,N,D]
__global__ __launch_bounds__(256) void build_awe_h(
    const float* __restrict__ w0, const float* __restrict__ w1,
    _Float16* __restrict__ awe, long long n)
{
  long long i = (long long)blockIdx.x * 256 + threadIdx.x;
  if (i >= n) return;
  long long b = i / ((long long)kN * kD);
  long long r = i % ((long long)kN * kD);
  float v = (r < (long long)kL * kD) ? w0[b * kL * kD + r]
                                     : w1[b * kL * kD + r - (long long)kL * kD];
  awe[i] = (_Float16)v;
}

// goal_trans[b,h] = node_hidden[b] . gW[h] + gb[h]; also L2 norm per batch
__global__ __launch_bounds__(256) void goal_kernel(
    const float* __restrict__ nh, const float* __restrict__ gW,
    const float* __restrict__ gb, float* __restrict__ ngraw,
    float* __restrict__ gnorm)
{
  __shared__ float sm[256];
  int b = blockIdx.x, h = threadIdx.x;
  const float* x = nh + (long long)b * kD;
  const float* w = gW + (long long)h * kD;
  float s = gb[h];
  for (int d = 0; d < kD; ++d) s += x[d] * w[d];
  ngraw[(long long)b * kH + h] = s;
  float ss = block_reduce_sum256(s * s, sm);
  if (h == 0) gnorm[b] = sqrtf(ss);
}

// word_g[b,n,h] = l2(goal_trans[b]*gw)[h] = ngraw*gw / (|gw|*||ngraw|| + eps)
__global__ __launch_bounds__(256) void build_g_h(
    const float* __restrict__ ngraw, const float* __restrict__ gnorm,
    const float* __restrict__ gwd, _Float16* __restrict__ gh, long long n)
{
  long long i = (long long)blockIdx.x * 256 + threadIdx.x;
  if (i >= n) return;
  int h = (int)(i % kH);
  long long bn = i / kH;
  int b = (int)(bn / kN);
  float gw  = gwd[bn];
  float v   = ngraw[(long long)b * kH + h] * gw;
  float nrm = fabsf(gw) * gnorm[b];
  gh[i] = (_Float16)(v / (nrm + kEPS));
}

// masked-mean row normalization (optionally elementwise product of 2 masks),
// one pass f32 -> normalized f16. One block per row, C columns.
__global__ __launch_bounds__(256) void rownorm_mask_f16(
    const float* __restrict__ in1, const float* __restrict__ in2,
    _Float16* __restrict__ out, int C)
{
  __shared__ float sm[256];
  long long row = blockIdx.x;
  const float* p1 = in1 + row * C;
  const float* p2 = in2 ? in2 + row * C : nullptr;
  float cnt = 0.0f;
  for (int c = threadIdx.x; c < C; c += 256) {
    float v = p1[c] * (p2 ? p2[c] : 1.0f);
    cnt += (v != 0.0f) ? 1.0f : 0.0f;
  }
  float tot = block_reduce_sum256(cnt, sm);
  float inv = 1.0f / (tot + kEPS);
  for (int c = threadIdx.x; c < C; c += 256) {
    float v = p1[c] * (p2 ? p2[c] : 1.0f);
    out[row * C + c] = (_Float16)(v * inv);
  }
}

// adj_op[b,p,n] = wes[b,n]*word_operator[b,n,p], row-normalized over n -> f16
__global__ __launch_bounds__(256) void adjn_f16(
    const float* __restrict__ wes, const float* __restrict__ wop,
    _Float16* __restrict__ out)
{
  __shared__ float sm[256];
  int bp = blockIdx.x;
  int b = bp / kP, p = bp % kP;
  float cnt = 0.0f;
  for (int n = threadIdx.x; n < kN; n += 256) {
    float v = wes[(long long)b * kN + n] * wop[((long long)b * kN + n) * kP + p];
    cnt += (v != 0.0f) ? 1.0f : 0.0f;
  }
  float tot = block_reduce_sum256(cnt, sm);
  float inv = 1.0f / (tot + kEPS);
  for (int n = threadIdx.x; n < kN; n += 256) {
    float v = wes[(long long)b * kN + n] * wop[((long long)b * kN + n) * kP + p];
    out[(long long)bp * kN + n] = (_Float16)(v * inv);
  }
}

// per-batch transpose: in f32 [R,Cc] (row stride Cc) -> out f16 [Cc,R]
__global__ __launch_bounds__(256) void transpose_f32_to_f16(
    const float* __restrict__ in, int R, int Cc, long long strideIn,
    _Float16* __restrict__ out, long long strideOut)
{
  __shared__ float tile[32][33];
  const float* inp = in + (long long)blockIdx.z * strideIn;
  _Float16*   outp = out + (long long)blockIdx.z * strideOut;
  int r0 = blockIdx.y * 32, c0 = blockIdx.x * 32;
  int tx = threadIdx.x & 31, ty = threadIdx.x >> 5;   // 32 x 8
  for (int i = 0; i < 32; i += 8)
    tile[ty + i][tx] = inp[(long long)(r0 + ty + i) * Cc + c0 + tx];
  __syncthreads();
  for (int i = 0; i < 32; i += 8)
    outp[(long long)(c0 + ty + i) * R + r0 + tx] = (_Float16)tile[tx][ty + i];
}

// L2-normalize 256-wide rows, f32 -> f16
__global__ __launch_bounds__(256) void l2rows_f16(
    const float* __restrict__ in, _Float16* __restrict__ out)
{
  __shared__ float sm[256];
  long long row = blockIdx.x;
  float v = in[row * kH + threadIdx.x];
  float ss = block_reduce_sum256(v * v, sm);
  float nrm = sqrtf(ss);
  out[row * kH + threadIdx.x] = (_Float16)(v / (nrm + kEPS));
}

// gated word update + scatter to outputs 0/1 + f16 copy for next GEMM
__global__ __launch_bounds__(256) void word_update(
    const float* __restrict__ fg, const float* __restrict__ wu,
    const float* __restrict__ w0, const float* __restrict__ w1,
    float* __restrict__ outp, _Float16* __restrict__ wu_h, long long n)
{
  long long i = (long long)blockIdx.x * 256 + threadIdx.x;
  if (i >= n) return;
  long long b = i / ((long long)kN * kD);
  long long r = i % ((long long)kN * kD);
  float awe = (r < (long long)kL * kD) ? w0[b * kL * kD + r]
                                       : w1[b * kL * kD + r - (long long)kL * kD];
  float f = 1.0f / (1.0f + __expf(-fg[i]));
  float u = fmaxf(wu[i], 0.0f);
  float res = fmaxf(f, kFT) * awe + (1.0f - f) * u;
  wu_h[i] = (_Float16)res;
  long long half0 = (long long)kB * kL * kD;
  long long o = (r < (long long)kL * kD)
      ? (b * kL * kD + r)
      : (half0 + b * kL * kD + (r - (long long)kL * kD));
  outp[o] = res;
}

// gated operator update -> output 2
__global__ __launch_bounds__(256) void op_update(
    const float* __restrict__ fg2, const float* __restrict__ lo,
    const float* __restrict__ ope, float* __restrict__ outp, long long n)
{
  long long i = (long long)blockIdx.x * 256 + threadIdx.x;
  if (i >= n) return;
  float f = 1.0f / (1.0f + __expf(-fg2[i]));
  float res = fmaxf(f, kFT) * ope[i] + (1.0f - f) * fmaxf(lo[i], 0.0f);
  outp[i] = res;
}

// ---------------------------------------------------------------------------
// Host orchestration
// ---------------------------------------------------------------------------
extern "C" void kernel_launch(void* const* d_in, const int* in_sizes, int n_in,
                              void* d_out, int out_size, void* d_ws, size_t ws_size,
                              hipStream_t stream) {
  (void)in_sizes; (void)n_in; (void)out_size; (void)ws_size;

  const float* wo0  = (const float*)d_in[0];
  const float* wo1  = (const float*)d_in[1];
  const float* nh   = (const float*)d_in[2];
  const float* ope  = (const float*)d_in[3];
  const float* wes  = (const float*)d_in[4];
  const float* wem  = (const float*)d_in[5];
  const float* wopi = (const float*)d_in[6];
  const float* wwm  = (const float*)d_in[7];
  const float* dr0  = (const float*)d_in[8];
  const float* dr1  = (const float*)d_in[9];
  const float* gwd  = (const float*)d_in[10];
  const float* gW   = (const float*)d_in[11];
  const float* gb   = (const float*)d_in[12];
  const float* wkW  = (const float*)d_in[13];
  const float* wkb  = (const float*)d_in[14];
  const float* wsW  = (const float*)d_in[15];
  const float* wsb  = (const float*)d_in[16];
  const float* woW  = (const float*)d_in[17];
  const float* wob  = (const float*)d_in[18];
  const float* fgW  = (const float*)d_in[19];
  const float* fgb  = (const float*)d_in[20];
  const float* wuW  = (const float*)d_in[21];
  const float* wub  = (const float*)d_in[22];
  const float* fg2W = (const float*)d_in[23];
  const float* fg2b = (const float*)d_in[24];
  const float* loW  = (const float*)d_in[25];
  const float* lob  = (const float*)d_in[26];

  const long long BND = (long long)kB * kN * kD;   // 8388608
  const long long BPD = (long long)kB * kP * kD;   // 262144

  uint8_t* base = (uint8_t*)d_ws;
  size_t off = 0;
  auto carve = [&](size_t bytes) -> void* {
    void* p = base + off;
    off += (bytes + 255) & ~(size_t)255;
    return p;
  };

  _Float16* awe_h  = (_Float16*)carve(BND * 2);
  _Float16* g_h    = (_Float16*)carve(BND * 2);
  _Float16* k_h    = (_Float16*)carve(BND * 2);
  _Float16* s_h    = (_Float16*)carve(BND * 2);
  _Float16* wuh    = (_Float16*)carve(BND * 2);
  _Float16* wk_hT  = (_Float16*)carve(BND * 2);
  _Float16* tT     = (_Float16*)carve(BND * 2);
  _Float16* lin0T  = (_Float16*)carve((long long)kB * kH * kL * 2);
  _Float16* lin1T  = (_Float16*)carve((long long)kB * kH * kL * 2);
  _Float16* Mh     = (_Float16*)carve((long long)kB * kN * kN * 2);
  _Float16* D0h    = (_Float16*)carve((long long)kB * kL * kL * 2);
  _Float16* D1h    = (_Float16*)carve((long long)kB * kL * kL * 2);
  _Float16* adjn   = (_Float16*)carve((long long)kB * kP * kN * 2);
  float*    F1     = (float*)carve(BND * 4);
  float*    F2     = (float*)carve(BND * 4);
  float*    ngraw  = (float*)carve((long long)kB * kH * 4);
  float*    gnorm  = (float*)carve(kB * 4);
  _Float16* wkW_h  = (_Float16*)carve(65536 * 2);
  _Float16* wsW_h  = (_Float16*)carve(65536 * 2);
  _Float16* woW_h  = (_Float16*)carve(65536 * 2);
  _Float16* fgW_h  = (_Float16*)carve(262144 * 2);
  _Float16* wuW_h  = (_Float16*)carve(196608 * 2);
  _Float16* fg2W_h = (_Float16*)carve(131072 * 2);
  _Float16* loW_h  = (_Float16*)carve(65536 * 2);
  _Float16* ope_h  = (_Float16*)carve(BPD * 2);
  _Float16* wop_h  = (_Float16*)carve(BPD * 2);
  float*    wopf   = (float*)carve(BPD * 4);
  float*    G1     = (float*)carve(BPD * 4);
  float*    G2     = (float*)carve(BPD * 4);

  dim3 blk(256);
  auto g1 = [](long long n) { return dim3((unsigned)((n + 255) / 256)); };

  // ---- conversions ----
  f32_to_f16<<<g1(65536),  blk, 0, stream>>>(wkW,  wkW_h,  65536);
  f32_to_f16<<<g1(65536),  blk, 0, stream>>>(wsW,  wsW_h,  65536);
  f32_to_f16<<<g1(65536),  blk, 0, stream>>>(woW,  woW_h,  65536);
  f32_to_f16<<<g1(262144), blk, 0, stream>>>(fgW,  fgW_h,  262144);
  f32_to_f16<<<g1(196608), blk, 0, stream>>>(wuW,  wuW_h,  196608);
  f32_to_f16<<<g1(131072), blk, 0, stream>>>(fg2W, fg2W_h, 131072);
  f32_to_f16<<<g1(65536),  blk, 0, stream>>>(loW,  loW_h,  65536);
  f32_to_f16<<<g1(BPD),    blk, 0, stream>>>(ope,  ope_h,  BPD);
  build_awe_h<<<g1(BND), blk, 0, stream>>>(wo0, wo1, awe_h, BND);

  // ---- goal message ----
  goal_kernel<<<dim3(kB), blk, 0, stream>>>(nh, gW, gb, ngraw, gnorm);
  build_g_h<<<g1(BND), blk, 0, stream>>>(ngraw, gnorm, gwd, g_h, BND);

  // ---- mask normalizations (one pass over the 134MB masks) ----
  rownorm_mask_f16<<<dim3(kB * kN), blk, 0, stream>>>(wwm, wem, Mh, kN);
  rownorm_mask_f16<<<dim3(kB * kL), blk, 0, stream>>>(dr0, nullptr, D0h, kL);
  rownorm_mask_f16<<<dim3(kB * kL), blk, 0, stream>>>(dr1, nullptr, D1h, kL);
  adjn_f16<<<dim3(kB * kP), blk, 0, stream>>>(wes, wopi, adjn);

  const long long sC = (long long)kN * kH;           // per-batch C stride (N*256)

  // ---- wk linear, then transpose to [H,N] per batch ----
  wmma_gemm<<<dim3(2, 256, 1), blk, 0, stream>>>(awe_h, kD, 0, wkW_h, kD, 0,
      F1, kH, 0, wkb, kB * kN, kD, 0);
  transpose_f32_to_f16<<<dim3(8, 32, kB), blk, 0, stream>>>(
      F1, kN, kH, sC, wk_hT, (long long)kH * kN);

  // ---- ws linear (both spans at once), transpose halves ----
  wmma_gemm<<<dim3(2, 256, 1), blk, 0, stream>>>(awe_h, kD, 0, wsW_h, kD, 0,
      F1, kH, 0, wsb, kB * kN, kD, 0);
  transpose_f32_to_f16<<<dim3(8, 16, kB), blk, 0, stream>>>(
      F1, kL, kH, sC, lin0T, (long long)kH * kL);
  transpose_f32_to_f16<<<dim3(8, 16, kB), blk, 0, stream>>>(
      F1 + (long long)kL * kH, kL, kH, sC, lin1T, (long long)kH * kL);

  // ---- knowledge message: K_raw = Mh @ wk, then L2 ----
  wmma_gemm<<<dim3(2, 8, kB), blk, 0, stream>>>(Mh, kN, (long long)kN * kN,
      wk_hT, kN, (long long)kH * kN, F2, kH, sC, nullptr, kN, kN, 0);
  l2rows_f16<<<dim3(kB * kN), blk, 0, stream>>>(F2, k_h);

  // ---- syntax message: S0/S1 = normD @ lin, then L2 ----
  wmma_gemm<<<dim3(2, 4, kB), blk, 0, stream>>>(D0h, kL, (long long)kL * kL,
      lin0T, kL, (long long)kH * kL, F1, kH, sC, nullptr, kL, kL, 0);
  wmma_gemm<<<dim3(2, 4, kB), blk, 0, stream>>>(D1h, kL, (long long)kL * kL,
      lin1T, kL, (long long)kH * kL, F1 + (long long)kL * kH, kH, sC,
      nullptr, kL, kL, 0);
  l2rows_f16<<<dim3(kB * kN), blk, 0, stream>>>(F1, s_h);

  // ---- forget gate: sum of 4 GEMMs over fgW column slices -> F1 ----
  wmma_gemm<<<dim3(2, 256, 1), blk, 0, stream>>>(awe_h, kD, 0, fgW_h + 0,   1024, 0, F1, kD, 0, fgb,     kB * kN, kD, 0);
  wmma_gemm<<<dim3(2, 256, 1), blk, 0, stream>>>(g_h,   kH, 0, fgW_h + 256, 1024, 0, F1, kD, 0, nullptr, kB * kN, kH, 1);
  wmma_gemm<<<dim3(2, 256, 1), blk, 0, stream>>>(k_h,   kH, 0, fgW_h + 512, 1024, 0, F1, kD, 0, nullptr, kB * kN, kH, 1);
  wmma_gemm<<<dim3(2, 256, 1), blk, 0, stream>>>(s_h,   kH, 0, fgW_h + 768, 1024, 0, F1, kD, 0, nullptr, kB * kN, kH, 1);
  // ---- update path: sum of 3 GEMMs over wuW column slices -> F2 ----
  wmma_gemm<<<dim3(2, 256, 1), blk, 0, stream>>>(g_h, kH, 0, wuW_h + 0,   768, 0, F2, kD, 0, wub,     kB * kN, kH, 0);
  wmma_gemm<<<dim3(2, 256, 1), blk, 0, stream>>>(k_h, kH, 0, wuW_h + 256, 768, 0, F2, kD, 0, nullptr, kB * kN, kH, 1);
  wmma_gemm<<<dim3(2, 256, 1), blk, 0, stream>>>(s_h, kH, 0, wuW_h + 512, 768, 0, F2, kD, 0, nullptr, kB * kN, kH, 1);

  word_update<<<g1(BND), blk, 0, stream>>>(F1, F2, wo0, wo1,
      (float*)d_out, wuh, BND);

  // ---- operator aggregation ----
  wmma_gemm<<<dim3(2, 256, 1), blk, 0, stream>>>(wuh, kD, 0, woW_h, kD, 0,
      F1, kD, 0, wob, kB * kN, kD, 0);
  transpose_f32_to_f16<<<dim3(8, 32, kB), blk, 0, stream>>>(
      F1, kN, kD, sC, tT, (long long)kD * kN);
  wmma_gemm<<<dim3(2, 1, kB), blk, 0, stream>>>(adjn, kN, (long long)kP * kN,
      tT, kN, (long long)kD * kN, wopf, kD, (long long)kP * kD,
      nullptr, kP, kN, 0);
  f32_to_f16<<<g1(BPD), blk, 0, stream>>>(wopf, wop_h, BPD);

  // ---- operator gate + update ----
  wmma_gemm<<<dim3(2, 8, 1), blk, 0, stream>>>(ope_h, kD, 0, fg2W_h + 0,   512, 0, G1, kD, 0, fg2b,    kB * kP, kD, 0);
  wmma_gemm<<<dim3(2, 8, 1), blk, 0, stream>>>(wop_h, kD, 0, fg2W_h + 256, 512, 0, G1, kD, 0, nullptr, kB * kP, kD, 1);
  wmma_gemm<<<dim3(2, 8, 1), blk, 0, stream>>>(wop_h, kD, 0, loW_h, kD, 0,
      G2, kD, 0, lob, kB * kP, kD, 0);

  op_update<<<g1(BPD), blk, 0, stream>>>(G1, G2, ope,
      (float*)d_out + 2LL * kB * kL * kD, BPD);
}
